// ProGenBase_26603027432166
// MI455X (gfx1250) — matrix operand order repaired and checked by reading, not compiled, not run on hardware
//
#include <hip/hip_runtime.h>

#define SEQ    2048
#define DIM    1024
#define VOCAB  512
#define HEADS  8
#define DHEAD  64
#define WSZ    256
#define NW     (SEQ / WSZ)        // 8 windows
#define INNER  (HEADS * DHEAD)    // 512
#define LN_EPS 1e-5f

// ---- feature gate: Tensor Data Mover tile loads ----
#ifndef USE_TDM
#define USE_TDM 1
#endif
#if defined(__has_builtin)
# if !__has_builtin(__builtin_amdgcn_tensor_load_to_lds)
#  undef USE_TDM
#  define USE_TDM 0
# endif
#endif

typedef unsigned short bh; // bf16 bit pattern
typedef __attribute__((ext_vector_type(16))) __bf16          v16bf;
typedef __attribute__((ext_vector_type(16))) unsigned short  v16us;
typedef __attribute__((ext_vector_type(8)))  float           v8f;
typedef __attribute__((ext_vector_type(4)))  unsigned int    u32x4v;
typedef __attribute__((ext_vector_type(8)))  int             i32x8v;
typedef __attribute__((ext_vector_type(4)))  int             i32x4v;

union BFrag { v16us u; v16bf b; u32x4v q[2]; };

__device__ __forceinline__ bh f2bf(float f) {
  unsigned int u = __float_as_uint(f);
  u += 0x7fffu + ((u >> 16) & 1u);          // round-to-nearest-even
  return (bh)(u >> 16);
}
__device__ __forceinline__ float gelu_f(float x) {
  float x3 = x * x * x;
  return 0.5f * x * (1.f + tanhf(0.7978845608028654f * (x + 0.044715f * x3)));
}

#if USE_TDM
// 2D TDM descriptor load: tile = 32 (K, dim0) x 64 (rows, dim1) bf16 elements,
// LDS rows padded 32->48 elements via pad_enable (interval 16 dwords, amount 8).
__device__ __forceinline__ void tdm_load_tile(unsigned lds_off, const bh* gptr,
                                              unsigned td0_rem, unsigned td1_rem,
                                              unsigned stride_elems)
{
  unsigned long long ga = (unsigned long long)gptr;
  u32x4v g0;
  g0[0] = 1u;                                 // count=1 (user descriptor)
  g0[1] = lds_off;                            // LDS byte address
  g0[2] = (unsigned)ga;                       // global addr [31:0]
  g0[3] = (unsigned)(ga >> 32) | (2u << 30);  // addr [56:32] | type=2 (image)
  i32x8v g1;
  g1[0] = (1 << 16)        // data_size = 1 -> 2 bytes
        | (1 << 20)        // pad_enable
        | (3 << 22)        // pad_interval: 16 dwords (= one 32-elem bf16 row)
        | (7 << 25);       // pad_amount: 8 dwords (= 16 bf16 elems of pad)
  g1[1] = (int)((td0_rem & 0xffffu) << 16);                    // tensor_dim0 lo
  g1[2] = (int)((td0_rem >> 16) | ((td1_rem & 0xffffu) << 16)); // dim0 hi | dim1 lo
  g1[3] = (int)((td1_rem >> 16) | (32u << 16));                 // dim1 hi | tile_dim0=32
  g1[4] = (int)64;                                              // tile_dim1=64, tile_dim2=0
  g1[5] = (int)stride_elems;                                    // tensor_dim0_stride lo
  g1[6] = 0;
  g1[7] = 0;
  i32x4v z4 = {0, 0, 0, 0};
#if __clang_major__ >= 23
  i32x8v z8 = {0, 0, 0, 0, 0, 0, 0, 0};
  __builtin_amdgcn_tensor_load_to_lds(g0, g1, z4, z4, z8, 0);
#else
  __builtin_amdgcn_tensor_load_to_lds(g0, g1, z4, z4, 0);
#endif
}
#endif

// ---------------------------------------------------------------------------
// Tiled bf16 WMMA GEMM: C[M,N] = A[M,K] @ Bt[N,K]^T (+bias[N]) (+residual)
// A row-major [M,K]; B pre-transposed K-major [N,K] -> identical 64x32 tiles.
// Block: 256 threads = 8 waves (4 M x 2 N), wave computes 16x32 (2x wmma).
// TDM path: double-buffered LDS tiles; wave 0 issues tensor_load_to_lds for
// tile t+1, waits tensorcnt<=2 (tile t's two DMAs retired, in-order), so WMMA
// on tile t overlaps the DMA of tile t+1.
// triK!=0: A is lower-triangular in K (n<=m) -> cap K loop at bM+64.
// ---------------------------------------------------------------------------
__global__ __launch_bounds__(256) void gemm_bf16_wmma(
    const bh* __restrict__ A, const bh* __restrict__ Bt, float* __restrict__ C,
    const float* __restrict__ bias, const float* __restrict__ residual,
    int M, int N, int K, int triK)
{
  __shared__ bh As[2][64][48];   // 32 K used + 16 pad; row stride 96B (16B mult)
  __shared__ bh Bs[2][64][48];

  const int tid  = threadIdx.x;
  const int lane = tid & 31, wv = tid >> 5;
  const int half = lane >> 4, l16 = lane & 15;
  const int wm = wv >> 1, wn = wv & 1;
  const int bM = blockIdx.y * 64, bN = blockIdx.x * 64;

  const int kEnd = triK ? ((bM + 64 < K) ? (bM + 64) : K) : K;
  const int nT   = kEnd >> 5;

  v8f c0 = {}; v8f c1 = {};
  const int ar  = wm * 16 + l16;   // A row for this lane
  const int bc0 = wn * 32 + l16;   // B column (row of Bt) for this lane, frag 0

#if USE_TDM
  if (tid == 0) {
    tdm_load_tile((unsigned)(size_t)(void*)&As[0][0][0],
                  A  + (size_t)bM * K, (unsigned)K, (unsigned)(M - bM), (unsigned)K);
    tdm_load_tile((unsigned)(size_t)(void*)&Bs[0][0][0],
                  Bt + (size_t)bN * K, (unsigned)K, (unsigned)(N - bN), (unsigned)K);
  }
  for (int t = 0; t < nT; ++t) {
    const int cur = t & 1;
    if (tid == 0) {
      if (t + 1 < nT) {                     // prefetch tile t+1 via TDM
        const int k0n = (t + 1) << 5;
        tdm_load_tile((unsigned)(size_t)(void*)&As[cur ^ 1][0][0],
                      A  + (size_t)bM * K + k0n, (unsigned)(K - k0n), (unsigned)(M - bM), (unsigned)K);
        tdm_load_tile((unsigned)(size_t)(void*)&Bs[cur ^ 1][0][0],
                      Bt + (size_t)bN * K + k0n, (unsigned)(K - k0n), (unsigned)(N - bN), (unsigned)K);
        __builtin_amdgcn_s_wait_tensorcnt(2);   // tile t complete; t+1 in flight
      } else {
        __builtin_amdgcn_s_wait_tensorcnt(0);
      }
    }
    __syncthreads();                        // tile t visible to all waves

    BFrag a, b0, b1;
    a.q[0]  = *(const u32x4v*)&As[cur][ar][half * 16];
    a.q[1]  = *(const u32x4v*)&As[cur][ar][half * 16 + 8];
    b0.q[0] = *(const u32x4v*)&Bs[cur][bc0][half * 16];
    b0.q[1] = *(const u32x4v*)&Bs[cur][bc0][half * 16 + 8];
    b1.q[0] = *(const u32x4v*)&Bs[cur][bc0 + 16][half * 16];
    b1.q[1] = *(const u32x4v*)&Bs[cur][bc0 + 16][half * 16 + 8];

    c0 = __builtin_amdgcn_wmma_f32_16x16x32_bf16(false, a.b, false, b0.b, (short)0, c0, false, false);
    c1 = __builtin_amdgcn_wmma_f32_16x16x32_bf16(false, a.b, false, b1.b, (short)0, c1, false, false);

    __syncthreads();                        // done reading buf cur before reuse
  }
#else
  const int trow = tid >> 2, tcol = (tid & 3) * 8;   // 64 rows x 32 K, 8 bh/thread
  for (int k0 = 0; k0 < kEnd; k0 += 32) {
    __syncthreads();
    *(u32x4v*)&As[0][trow][tcol] = *(const u32x4v*)(A  + (size_t)(bM + trow) * K + k0 + tcol);
    *(u32x4v*)&Bs[0][trow][tcol] = *(const u32x4v*)(Bt + (size_t)(bN + trow) * K + k0 + tcol);
    if (k0 + 32 < kEnd) {                    // speculative prefetch of next tiles
      __builtin_prefetch(A  + (size_t)(bM + trow) * K + k0 + 32 + tcol, 0, 0);
      __builtin_prefetch(Bt + (size_t)(bN + trow) * K + k0 + 32 + tcol, 0, 0);
    }
    __syncthreads();

    BFrag a, b0, b1;
    a.q[0]  = *(const u32x4v*)&As[0][ar][half * 16];
    a.q[1]  = *(const u32x4v*)&As[0][ar][half * 16 + 8];
    b0.q[0] = *(const u32x4v*)&Bs[0][bc0][half * 16];
    b0.q[1] = *(const u32x4v*)&Bs[0][bc0][half * 16 + 8];
    b1.q[0] = *(const u32x4v*)&Bs[0][bc0 + 16][half * 16];
    b1.q[1] = *(const u32x4v*)&Bs[0][bc0 + 16][half * 16 + 8];

    c0 = __builtin_amdgcn_wmma_f32_16x16x32_bf16(false, a.b, false, b0.b, (short)0, c0, false, false);
    c1 = __builtin_amdgcn_wmma_f32_16x16x32_bf16(false, a.b, false, b1.b, (short)0, c1, false, false);
  }
#endif

#pragma unroll
  for (int v = 0; v < 8; ++v) {
    int row = bM + wm * 16 + half * 8 + v;   // C/D: vgpr v -> M=v (+8 upper half)
    int col = bN + wn * 32 + l16;
    float r0 = c0[v], r1 = c1[v];
    if (bias)     { r0 += bias[col];                      r1 += bias[col + 16]; }
    if (residual) { r0 += residual[(size_t)row*N + col];  r1 += residual[(size_t)row*N + col + 16]; }
    C[(size_t)row*N + col]      = r0;
    C[(size_t)row*N + col + 16] = r1;
  }
}

// ---------------------------------------------------------------------------
// Windowed attention scores: sim[h,w,i,j] = scale * q_hw[i,:] . k_cat[j,:]
// q,k row-major [head][pos][DHEAD] -> all fragments are global b128 loads.
// ---------------------------------------------------------------------------
__global__ __launch_bounds__(256) void attn_scores(
    const bh* __restrict__ q, const bh* __restrict__ k, float* __restrict__ sim)
{
  const int gid  = blockIdx.x * 8 + (threadIdx.x >> 5);  // 0 .. 64*512-1
  const int lane = threadIdx.x & 31;
  const int half = lane >> 4, l16 = lane & 15;
  const int hw = gid >> 9;            // head*NW + window
  const int t  = gid & 511;
  const int mt = t >> 5, nt = t & 31; // 16 M-tiles x 32 N-tiles
  const int h = hw >> 3, w = hw & 7;

  const bh* qrow = q + ((size_t)h * SEQ + w * WSZ + mt * 16 + l16) * DHEAD;
  const int j    = nt * 16 + l16;
  const int kpos = (w - 1) * WSZ + j;
  const bh* krow = (kpos >= 0) ? (k + ((size_t)h * SEQ + kpos) * DHEAD) : nullptr;

  v8f c = {};
#pragma unroll
  for (int ks = 0; ks < DHEAD; ks += 32) {
    BFrag a, b;
    a.q[0] = *(const u32x4v*)(qrow + ks + half * 16);
    a.q[1] = *(const u32x4v*)(qrow + ks + half * 16 + 8);
    if (krow) {
      b.q[0] = *(const u32x4v*)(krow + ks + half * 16);
      b.q[1] = *(const u32x4v*)(krow + ks + half * 16 + 8);
    } else {
      b.q[0] = (u32x4v){0,0,0,0}; b.q[1] = (u32x4v){0,0,0,0};
    }
    c = __builtin_amdgcn_wmma_f32_16x16x32_bf16(false, a.b, false, b.b, (short)0, c, false, false);
  }
#pragma unroll
  for (int v = 0; v < 8; ++v) {
    int i = mt * 16 + half * 8 + v;
    float val = (j <= i + WSZ) ? c[v] * 0.125f : -1.0e10f;   // scale = DHEAD^-0.5
    sim[((size_t)hw * WSZ + i) * (2 * WSZ) + j] = val;
  }
}

// attn @ v : v stored head-transposed vT[head][d][pos] so B frags are b128.
__global__ __launch_bounds__(256) void attn_av(
    const bh* __restrict__ probs, const bh* __restrict__ vT, bh* __restrict__ out)
{
  const int gid  = blockIdx.x * 8 + (threadIdx.x >> 5);  // 0 .. 64*64-1
  const int lane = threadIdx.x & 31;
  const int half = lane >> 4, l16 = lane & 15;
  const int hw = gid >> 6;
  const int t  = gid & 63;
  const int mt = t >> 2, nt = t & 3;
  const int h = hw >> 3, w = hw & 7;

  const bh* prow = probs + ((size_t)hw * WSZ + mt * 16 + l16) * (2 * WSZ);
  const int d    = nt * 16 + l16;
  const bh* vrow = vT + ((size_t)h * DHEAD + d) * SEQ;   // positions contiguous

  v8f c = {};
  for (int ks = 0; ks < 2 * WSZ; ks += 32) {
    BFrag a, b;
    a.q[0] = *(const u32x4v*)(prow + ks + half * 16);
    a.q[1] = *(const u32x4v*)(prow + ks + half * 16 + 8);
    const int vpos0 = (w - 1) * WSZ + ks + half * 16;      // 16-aligned: no straddle
    if (vpos0 >= 0) {
      b.q[0] = *(const u32x4v*)(vrow + vpos0);
      b.q[1] = *(const u32x4v*)(vrow + vpos0 + 8);
      __builtin_prefetch(vrow + vpos0 + 32, 0, 0);         // global_prefetch_b8
    } else {
      b.q[0] = (u32x4v){0,0,0,0}; b.q[1] = (u32x4v){0,0,0,0};
    }
    c = __builtin_amdgcn_wmma_f32_16x16x32_bf16(false, a.b, false, b.b, (short)0, c, false, false);
  }
#pragma unroll
  for (int v = 0; v < 8; ++v) {
    int pos = w * WSZ + mt * 16 + half * 8 + v;
    out[(size_t)pos * INNER + h * DHEAD + d] = f2bf(c[v]);   // [pos][h*64+d]
  }
}

// ---------------------------------------------------------------------------
// LayerNorm (fp32 in, scale, bf16 out). tstride==0: row-major out [row][W];
// tstride>0: transposed out [c][tstride] (K-major B operand for WMMA GEMM).
// ---------------------------------------------------------------------------
__global__ __launch_bounds__(256) void ln_bf16_kernel(
    const float* __restrict__ x, const float* __restrict__ g, bh* __restrict__ y,
    int W, int tstride)
{
  __shared__ float red[256];
  const int row = blockIdx.x, t = threadIdx.x;
  const float* xr = x + (size_t)row * W;
  float s = 0.f;
  for (int c = t; c < W; c += 256) s += xr[c];
  red[t] = s; __syncthreads();
  for (int o = 128; o > 0; o >>= 1) { if (t < o) red[t] += red[t + o]; __syncthreads(); }
  float mu = red[0] / (float)W;
  __syncthreads();
  float vv = 0.f;
  for (int c = t; c < W; c += 256) { float d = xr[c] - mu; vv += d * d; }
  red[t] = vv; __syncthreads();
  for (int o = 128; o > 0; o >>= 1) { if (t < o) red[t] += red[t + o]; __syncthreads(); }
  float rstd = rsqrtf(red[0] / (float)W + LN_EPS);
  if (tstride == 0) {
    bh* yr = y + (size_t)row * W;
    for (int c = t; c < W; c += 256) yr[c] = f2bf((xr[c] - mu) * rstd * g[c]);
  } else {
    for (int c = t; c < W; c += 256) y[(size_t)c * tstride + row] = f2bf((xr[c] - mu) * rstd * g[c]);
  }
}

// softmax over 512-wide rows of sim -> bf16 probabilities
__global__ __launch_bounds__(256) void softmax_rows(
    const float* __restrict__ sim, bh* __restrict__ probs)
{
  __shared__ float red[256];
  const int row = blockIdx.x, t = threadIdx.x;
  const float* s = sim + (size_t)row * (2 * WSZ);
  float a0 = s[t], a1 = s[t + 256];
  red[t] = fmaxf(a0, a1); __syncthreads();
  for (int o = 128; o > 0; o >>= 1) { if (t < o) red[t] = fmaxf(red[t], red[t+o]); __syncthreads(); }
  float mx = red[0]; __syncthreads();
  float e0 = expf(a0 - mx), e1 = expf(a1 - mx);
  red[t] = e0 + e1; __syncthreads();
  for (int o = 128; o > 0; o >>= 1) { if (t < o) red[t] += red[t+o]; __syncthreads(); }
  float inv = 1.f / red[0];
  bh* pr = probs + (size_t)row * (2 * WSZ);
  pr[t] = f2bf(e0 * inv); pr[t + 256] = f2bf(e1 * inv);
}

// ---------------------------------------------------------------------------
// Elementwise / data-movement helpers
// ---------------------------------------------------------------------------
__global__ void cast_f32_bf16(const float* __restrict__ in, bh* __restrict__ out, long n) {
  long i = (long)blockIdx.x * 256 + threadIdx.x;
  if (i < n) out[i] = f2bf(in[i]);
}

// tiled transpose + cast: in fp32 [K][N] -> out bf16 [N][K]
__global__ __launch_bounds__(256) void transpose_cast(
    const float* __restrict__ in, bh* __restrict__ out, int K, int N)
{
  __shared__ float tile[32][33];
  const int bk = blockIdx.y * 32, bn = blockIdx.x * 32;
  const int lx = threadIdx.x & 31, ly = threadIdx.x >> 5;   // 8 rows/pass
  for (int r = ly; r < 32; r += 8) tile[r][lx] = in[(size_t)(bk + r) * N + bn + lx];
  __syncthreads();
  for (int r = ly; r < 32; r += 8) out[(size_t)(bn + r) * K + bk + lx] = f2bf(tile[lx][r]);
}

__global__ void embed_gather(const float* __restrict__ emb, const int* __restrict__ tok,
                             float* __restrict__ x) {
  long i = (long)blockIdx.x * 256 + threadIdx.x;     // SEQ*DIM exact
  int row = (int)(i >> 10), c = (int)(i & 1023);
  x[i] = emb[(size_t)tok[row] * DIM + c];
}

// rotary on q,k,v (reference applies rotary to all three).
// q,k -> [head][pos][DHEAD];  v -> transposed [head][d][pos]
__global__ void rotary_split(const float* __restrict__ qkv, bh* __restrict__ q,
                             bh* __restrict__ k, bh* __restrict__ vT) {
  int idx = blockIdx.x * 256 + threadIdx.x;          // SEQ*HEADS*32 exact
  int i   = idx & 31;
  int h   = (idx >> 5) & 7;
  int pos = idx >> 8;
  float freq = powf(10000.f, -(float)i / 32.f);
  float ang = (float)pos * freq;
  float sn = sinf(ang), cs = cosf(ang);
  const float* row = qkv + (size_t)pos * (3 * INNER);
#pragma unroll
  for (int t = 0; t < 3; ++t) {
    float x0 = row[t * INNER + h * DHEAD + 2 * i];
    float x1 = row[t * INNER + h * DHEAD + 2 * i + 1];
    float r0 = x0 * cs - x1 * sn;
    float r1 = x1 * cs + x0 * sn;
    if (t < 2) {
      bh* o = (t == 0 ? q : k) + ((size_t)h * SEQ + pos) * DHEAD + 2 * i;
      o[0] = f2bf(r0); o[1] = f2bf(r1);
    } else {
      vT[((size_t)h * DHEAD + 2 * i) * SEQ + pos]     = f2bf(r0);
      vT[((size_t)h * DHEAD + 2 * i + 1) * SEQ + pos] = f2bf(r1);
    }
  }
}

// GLU: out = bf16(a * gelu(g)), h = [SEQ,8192] -> out [SEQ,4096]
__global__ void glu_act(const float* __restrict__ h, bh* __restrict__ out) {
  long i = (long)blockIdx.x * 256 + threadIdx.x;     // SEQ*4096 exact
  long row = i >> 12; int c = (int)(i & 4095);
  float a = h[row * 8192 + c];
  float g = h[row * 8192 + 4096 + c];
  out[i] = f2bf(a * gelu_f(g));
}

// SGU: gelu both halves of [SEQ,4096] -> a [SEQ,2048], gate [SEQ,2048]
__global__ void gelu_split(const float* __restrict__ h, float* __restrict__ a,
                           float* __restrict__ g) {
  long i = (long)blockIdx.x * 256 + threadIdx.x;     // SEQ*2048 exact
  long row = i >> 11; int c = (int)(i & 2047);
  a[i] = gelu_f(h[row * 4096 + c]);
  g[i] = gelu_f(h[row * 4096 + 2048 + c]);
}

// masked spatial weight (A operand, row-major): out[m,n] = bf16(w[m,n]) if n<=m
__global__ void mask_tril(const float* __restrict__ w, bh* __restrict__ out) {
  long i = (long)blockIdx.x * 256 + threadIdx.x;     // SEQ*SEQ exact
  int m = (int)(i >> 11), n = (int)(i & 2047);
  out[i] = (n <= m) ? f2bf(w[i]) : (bh)0;
}

// out = bf16(a * (gate2 + sgu_b[row]))
__global__ void sgu_combine(const float* __restrict__ a, const float* __restrict__ gate,
                            const float* __restrict__ b, bh* __restrict__ out) {
  long i = (long)blockIdx.x * 256 + threadIdx.x;     // SEQ*2048 exact
  long row = i >> 11;
  out[i] = f2bf(a[i] * (gate[i] + b[row]));
}

// ---------------------------------------------------------------------------
// Host orchestration
// ---------------------------------------------------------------------------
extern "C" void kernel_launch(void* const* d_in, const int* in_sizes, int n_in,
                              void* d_out, int out_size, void* d_ws, size_t ws_size,
                              hipStream_t stream)
{
  (void)in_sizes; (void)n_in; (void)out_size; (void)ws_size;

  // ---- inputs in setup_inputs() insertion order ----
  int p = 0;
  auto F = [&](void) { return (const float*)d_in[p++]; };
  const float* embed = F();
  struct LP {
    const float *a_ln, *a_wqkv, *a_wout, *a_bout;
    const float *f_ln, *f_win, *f_bin;
    const float *s_ln, *s_w, *s_b, *s_pw, *s_pb;
    const float *f_wout, *f_bout;
    bool sgu;
  } L[6];
  for (int i = 0; i < 6; ++i) {
    L[i].a_ln = F(); L[i].a_wqkv = F(); L[i].a_wout = F(); L[i].a_bout = F();
    L[i].f_ln = F(); L[i].f_win = F(); L[i].f_bin = F();
    L[i].sgu = (i >= 4);
    if (L[i].sgu) { L[i].s_ln = F(); L[i].s_w = F(); L[i].s_b = F(); L[i].s_pw = F(); L[i].s_pb = F(); }
    else          { L[i].s_ln = L[i].s_w = L[i].s_b = L[i].s_pw = L[i].s_pb = nullptr; }
    L[i].f_wout = F(); L[i].f_bout = F();
  }
  const float* final_ln = F();
  const float* logits_w = F();
  const float* logits_b = F();
  const int*   tokens   = (const int*)d_in[p++];

  // ---- workspace carve (256B aligned) ----
  char* base = (char*)d_ws; size_t off = 0;
  auto carve = [&](size_t bytes) -> void* {
    void* r = base + off; off += (bytes + 255) & ~(size_t)255; return r;
  };
  float* x     = (float*)carve((size_t)SEQ * DIM * 4);         // residual stream
  float* tmpF  = (float*)carve((size_t)SEQ * 8192 * 4);        // GEMM outputs
  float* tmpF2 = (float*)carve((size_t)SEQ * 4096 * 4);        // sgu a / gate
  bh*    lnB   = (bh*)carve((size_t)SEQ * 2048 * 2);           // LN outputs (bf16)
  bh*    actB  = (bh*)carve((size_t)SEQ * 4096 * 2);           // bf16 activations
  bh*    wB    = (bh*)carve((size_t)1024 * 8192 * 2);          // bf16 weights (K-major)
  bh*    qB    = (bh*)carve((size_t)HEADS * SEQ * DHEAD * 2);
  bh*    kB    = (bh*)carve((size_t)HEADS * SEQ * DHEAD * 2);
  bh*    vB    = (bh*)carve((size_t)HEADS * SEQ * DHEAD * 2);  // head-transposed
  float* sim   = (float*)carve((size_t)HEADS * NW * WSZ * 2 * WSZ * 4);
  bh*    probs = (bh*)carve((size_t)HEADS * NW * WSZ * 2 * WSZ * 2);
  bh*    attnO = (bh*)carve((size_t)SEQ * INNER * 2);

  auto cast = [&](const float* src, bh* dst, long n) {
    cast_f32_bf16<<<dim3((unsigned)((n + 255) / 256)), dim3(256), 0, stream>>>(src, dst, n);
  };
  auto castT = [&](const float* src, bh* dst, int K, int N) {   // [K][N] -> [N][K]
    transpose_cast<<<dim3(N / 32, K / 32), dim3(256), 0, stream>>>(src, dst, K, N);
  };
  auto gemm = [&](const bh* A, const bh* Bt, float* C, int M, int N, int K,
                  const float* bias, const float* res, int triK = 0) {
    gemm_bf16_wmma<<<dim3(N / 64, M / 64), dim3(256), 0, stream>>>(A, Bt, C, bias, res, M, N, K, triK);
  };

  // ---- embedding ----
  embed_gather<<<SEQ * DIM / 256, 256, 0, stream>>>(embed, tokens, x);

  for (int i = 0; i < 6; ++i) {
    // ---------- local attention ----------
    ln_bf16_kernel<<<SEQ, 256, 0, stream>>>(x, L[i].a_ln, lnB, DIM, 0);
    castT(L[i].a_wqkv, wB, DIM, 3 * INNER);
    gemm(lnB, wB, tmpF, SEQ, 3 * INNER, DIM, nullptr, nullptr);
    rotary_split<<<SEQ * HEADS * 32 / 256, 256, 0, stream>>>(tmpF, qB, kB, vB);
    attn_scores<<<HEADS * NW * 512 / 8, 256, 0, stream>>>(qB, kB, sim);
    softmax_rows<<<HEADS * NW * WSZ, 256, 0, stream>>>(sim, probs);
    attn_av<<<HEADS * NW * 64 / 8, 256, 0, stream>>>(probs, vB, attnO);
    castT(L[i].a_wout, wB, INNER, DIM);
    gemm(attnO, wB, x, SEQ, DIM, INNER, L[i].a_bout, x);   // + bias + residual

    if (!L[i].sgu) {
      // ---------- GLU feed-forward ----------
      ln_bf16_kernel<<<SEQ, 256, 0, stream>>>(x, L[i].f_ln, lnB, DIM, 0);
      castT(L[i].f_win, wB, DIM, 8192);
      gemm(lnB, wB, tmpF, SEQ, 8192, DIM, L[i].f_bin, nullptr);
      glu_act<<<SEQ * 4096 / 256, 256, 0, stream>>>(tmpF, actB);
      castT(L[i].f_wout, wB, 4096, DIM);
      gemm(actB, wB, x, SEQ, DIM, 4096, L[i].f_bout, x);
    } else {
      // ---------- SGU (gMLP) feed-forward ----------
      ln_bf16_kernel<<<SEQ, 256, 0, stream>>>(x, L[i].f_ln, lnB, DIM, 0);
      castT(L[i].f_win, wB, DIM, 4096);
      gemm(lnB, wB, tmpF, SEQ, 4096, DIM, L[i].f_bin, nullptr);
      gelu_split<<<SEQ * 2048 / 256, 256, 0, stream>>>(tmpF, tmpF2, tmpF2 + (size_t)SEQ * 2048);
      // gate LN written K-major (transposed) as B operand of the spatial GEMM
      ln_bf16_kernel<<<SEQ, 256, 0, stream>>>(tmpF2 + (size_t)SEQ * 2048, L[i].s_ln, lnB, 2048, SEQ);
      mask_tril<<<SEQ * SEQ / 256, 256, 0, stream>>>(L[i].s_w, wB);
      gemm(wB, lnB, tmpF, SEQ, 2048, SEQ, nullptr, nullptr, /*triK=*/1);  // spatial GEMM
      sgu_combine<<<SEQ * 2048 / 256, 256, 0, stream>>>(tmpF2, tmpF, L[i].s_b, actB);
      castT(L[i].s_pw, wB, 2048, 2048);
      gemm(actB, wB, tmpF, SEQ, 2048, 2048, L[i].s_pb, nullptr);
      cast(tmpF, actB, (long)SEQ * 2048);
      castT(L[i].f_wout, wB, 2048, DIM);
      gemm(actB, wB, x, SEQ, DIM, 2048, L[i].f_bout, x);
    }
  }

  // ---- final LN + logits ----
  ln_bf16_kernel<<<SEQ, 256, 0, stream>>>(x, final_ln, lnB, DIM, 0);
  castT(logits_w, wB, DIM, VOCAB);
  gemm(lnB, wB, (float*)d_out, SEQ, VOCAB, DIM, logits_b, nullptr);
}
